// KeyNet_59158879535448
// MI455X (gfx1250) — matrix operand order repaired and checked by reading, not compiled, not run on hardware
//
#include <hip/hip_runtime.h>
#include <cstdint>
#include <cstddef>

// ---------------- CDNA5 types ----------------
typedef __attribute__((ext_vector_type(16))) __bf16 v16bf;
typedef __attribute__((ext_vector_type(8)))  float  v8f;
typedef __attribute__((ext_vector_type(4)))  unsigned int u32x4;

__device__ __forceinline__ unsigned short f2bf(float f) {
  union { float f; unsigned int u; } c; c.f = f;
  unsigned int u = c.u;
  unsigned int r = u + 0x7FFFu + ((u >> 16) & 1u);   // round-to-nearest-even
  return (unsigned short)(r >> 16);
}

// ---- CDNA5 async copy: global -> LDS, tracked by ASYNCcnt ----
__device__ __forceinline__ void async_copy_b128(unsigned lds_off, unsigned long long gaddr) {
  asm volatile("global_load_async_to_lds_b128 %0, %1, off"
               :: "v"(lds_off), "v"(gaddr) : "memory");
}
__device__ __forceinline__ void wait_async0() {
  asm volatile("s_wait_asynccnt 0x0" ::: "memory");
}

// ---------------- center / v = pts - mean ----------------
__global__ void k_center_v(const float* __restrict__ pts, float* __restrict__ center,
                           float* __restrict__ V) {
  int b = blockIdx.x, t = threadIdx.x;
  __shared__ float sx[256], sy[256], sz[256];
  __shared__ float cx, cy, cz;
  float x = 0.f, y = 0.f, z = 0.f;
  for (int i = t; i < 4096; i += 256) {
    const float* p = pts + ((size_t)b * 4096 + i) * 3;
    x += p[0]; y += p[1]; z += p[2];
  }
  sx[t] = x; sy[t] = y; sz[t] = z; __syncthreads();
  for (int o = 128; o > 0; o >>= 1) {
    if (t < o) { sx[t] += sx[t+o]; sy[t] += sy[t+o]; sz[t] += sz[t+o]; }
    __syncthreads();
  }
  if (t == 0) {
    cx = sx[0] * (1.f/4096.f); cy = sy[0] * (1.f/4096.f); cz = sz[0] * (1.f/4096.f);
    center[b*3+0] = cx; center[b*3+1] = cy; center[b*3+2] = cz;
  }
  __syncthreads();
  for (int i = t; i < 4096; i += 256) {
    size_t r = ((size_t)b * 4096 + i) * 3;
    V[r+0] = pts[r+0] - cx; V[r+1] = pts[r+1] - cy; V[r+2] = pts[r+2] - cz;
  }
}

// ---------------- normalize direction columns (3 x m) ----------------
__global__ void k_normdir(const float* __restrict__ d, float* __restrict__ out, int m) {
  int j = blockIdx.x * blockDim.x + threadIdx.x;
  if (j >= m) return;
  float x = d[j], y = d[m + j], z = d[2*m + j];
  float n = fmaxf(sqrtf(x*x + y*y + z*z), 1e-12f);
  out[j] = x / n; out[m + j] = y / n; out[2*m + j] = z / n;
}

// ---------------- brute-force kNN (excludes self) ----------------
template<int KN>
__global__ void k_knn(const float* __restrict__ V, int nQ, int nC,
                      int* __restrict__ out, int total) {
  int q = blockIdx.x * blockDim.x + threadIdx.x;
  if (q >= total) return;
  int b = q / nQ, qi = q % nQ;
  const float* vb = V + (size_t)b * 4096 * 3;
  float qx = vb[qi*3], qy = vb[qi*3+1], qz = vb[qi*3+2];
  float bd[KN]; int bi[KN];
#pragma unroll
  for (int j = 0; j < KN; ++j) { bd[j] = 3.0e38f; bi[j] = 0; }
  float worst = 3.0e38f; int ws = 0;
  for (int c = 0; c < nC; ++c) {
    if (c == qi) continue;
    float dx = vb[c*3] - qx, dy = vb[c*3+1] - qy, dz = vb[c*3+2] - qz;
    float d = dx*dx + dy*dy + dz*dz;
    if (d < worst) {
#pragma unroll
      for (int j = 0; j < KN; ++j) if (j == ws) { bd[j] = d; bi[j] = c; }
      worst = -1.0f;
#pragma unroll
      for (int j = 0; j < KN; ++j) if (bd[j] > worst) { worst = bd[j]; ws = j; }
    }
  }
#pragma unroll
  for (int j = 0; j < KN; ++j) out[(size_t)q * KN + j] = bi[j];
}

// ---------------- nearest (argmin), 8192 queries ----------------
__global__ void k_nearest(const float* __restrict__ V, int nC, int* __restrict__ out) {
  int q = blockIdx.x * blockDim.x + threadIdx.x;
  if (q >= 8192) return;
  int b = q >> 12, qi = q & 4095;
  const float* vb = V + (size_t)b * 4096 * 3;
  float qx = vb[qi*3], qy = vb[qi*3+1], qz = vb[qi*3+2];
  float best = 3.0e38f; int bi2 = 0;
  for (int c = 0; c < nC; ++c) {
    float dx = vb[c*3] - qx, dy = vb[c*3+1] - qy, dz = vb[c*3+2] - qz;
    float d = dx*dx + dy*dy + dz*dz;
    if (d < best) { best = d; bi2 = c; }
  }
  out[q] = bi2;
}

// ---------------- conv_surface: fm0 (128 ch) ----------------
__global__ void k_conv_surface(const float* __restrict__ V, const int* __restrict__ idx,
                               const float* __restrict__ sd, float* __restrict__ out,
                               int nPerB) {
  int p = blockIdx.x;
  int b = p / nPerB, qi = p % nPerB;
  __shared__ float nd[10][3];
  const float* vb = V + (size_t)b * 4096 * 3;
  if (threadIdx.x < 10) {
    int nb = idx[(size_t)p * 10 + threadIdx.x];
    float dx = vb[nb*3]   - vb[qi*3];
    float dy = vb[nb*3+1] - vb[qi*3+1];
    float dz = vb[nb*3+2] - vb[qi*3+2];
    float n = fmaxf(sqrtf(dx*dx + dy*dy + dz*dz), 1e-12f);
    nd[threadIdx.x][0] = dx/n; nd[threadIdx.x][1] = dy/n; nd[threadIdx.x][2] = dz/n;
  }
  __syncthreads();
  int c = threadIdx.x;  // 128 threads
  float acc = 0.f;
  for (int s = 0; s < 7; ++s) {
    int m = s * 128 + c;
    float s0 = sd[m], s1 = sd[896 + m], s2 = sd[2*896 + m];
    float mx = 0.f;
#pragma unroll
    for (int k = 0; k < 10; ++k) {
      float t = nd[k][0]*s0 + nd[k][1]*s1 + nd[k][2]*s2;
      mx = fmaxf(mx, fmaxf(t, 0.f));
    }
    acc += mx;
  }
  out[(size_t)p * 128 + c] = fmaxf(acc, 0.f);
}

// ---------------- conv_layer aggregation: center + sum_s max_k(theta*supp) ----------------
__global__ void k_conv_agg(const float* __restrict__ V, const int* __restrict__ idx,
                           const float* __restrict__ sd, const float* __restrict__ fo,
                           float* __restrict__ out, int nPerB, int outC) {
  int p = blockIdx.x;
  int b = p / nPerB, qi = p % nPerB;
  __shared__ float nd[10][3];
  const float* vb = V + (size_t)b * 4096 * 3;
  if (threadIdx.x < 10) {
    int nb = idx[(size_t)p * 10 + threadIdx.x];
    float dx = vb[nb*3]   - vb[qi*3];
    float dy = vb[nb*3+1] - vb[qi*3+1];
    float dz = vb[nb*3+2] - vb[qi*3+2];
    float n = fmaxf(sqrtf(dx*dx + dy*dy + dz*dz), 1e-12f);
    nd[threadIdx.x][0] = dx/n; nd[threadIdx.x][1] = dy/n; nd[threadIdx.x][2] = dz/n;
  }
  __syncthreads();
  int c = threadIdx.x;        // blockDim.x == outC
  int foW = outC * 8;
  int thW = outC * 7;
  int nbr[10];
#pragma unroll
  for (int k = 0; k < 10; ++k) nbr[k] = idx[(size_t)p * 10 + k];
  float acc = fo[(size_t)p * foW + c];   // center part
  for (int s = 0; s < 7; ++s) {
    int m = s * outC + c;
    float s0 = sd[m], s1 = sd[thW + m], s2 = sd[2*thW + m];
    float mx = -3.0e38f;
#pragma unroll
    for (int k = 0; k < 10; ++k) {
      float t = fmaxf(nd[k][0]*s0 + nd[k][1]*s1 + nd[k][2]*s2, 0.f);
      float sup = fo[((size_t)b * nPerB + nbr[k]) * foW + (size_t)(s+1) * outC + c];
      mx = fmaxf(mx, t * sup);
    }
    acc += mx;
  }
  out[(size_t)p * outC + c] = acc;
}

// ---------------- BN stats (mu, rsqrt(var+eps)) ----------------
__global__ void k_bn_stats(const float* __restrict__ x, float* __restrict__ stat,
                           int M, int C) {
  int c = blockIdx.x;
  __shared__ float ss[256], ss2[256];
  float s = 0.f, s2 = 0.f;
  for (int i = threadIdx.x; i < M; i += 256) {
    float v = x[(size_t)i * C + c];
    s += v; s2 += v * v;
  }
  ss[threadIdx.x] = s; ss2[threadIdx.x] = s2; __syncthreads();
  for (int o = 128; o > 0; o >>= 1) {
    if (threadIdx.x < o) { ss[threadIdx.x] += ss[threadIdx.x+o]; ss2[threadIdx.x] += ss2[threadIdx.x+o]; }
    __syncthreads();
  }
  if (threadIdx.x == 0) {
    float mu = ss[0] / (float)M;
    float var = ss2[0] / (float)M - mu * mu;
    stat[c] = mu;
    stat[512 + c] = rsqrtf(var + 1e-5f);
  }
}

__global__ void k_bn_apply(const float* __restrict__ x, float* __restrict__ y,
                           const float* __restrict__ g, const float* __restrict__ bb,
                           const float* __restrict__ stat, size_t total, int C) {
  size_t id = (size_t)blockIdx.x * blockDim.x + threadIdx.x;
  if (id >= total) return;
  int c = (int)(id % C);
  float v = (x[id] - stat[c]) * stat[512 + c] * g[c] + bb[c];
  y[id] = fmaxf(v, 0.f);
}

// ---------------- pool gather-max over 4 neighbors ----------------
__global__ void k_gather_max(const float* __restrict__ f, const int* __restrict__ idx,
                             float* __restrict__ out, int C, int nOutPerB, int candPerB) {
  int q = blockIdx.x;
  int b = q / nOutPerB;
  int c = threadIdx.x;
  float mx = -3.0e38f;
#pragma unroll
  for (int j = 0; j < 4; ++j) {
    int nb = idx[(size_t)q * 4 + j];
    mx = fmaxf(mx, f[((size_t)b * candPerB + nb) * C + c]);
  }
  out[(size_t)q * C + c] = mx;
}

// ---------------- global max over 256 points (fm4 -> f_global) ----------------
__global__ void k_global_max(const float* __restrict__ fm4, float* __restrict__ fg) {
  int b = blockIdx.x, c = threadIdx.x;  // 512 threads
  float mx = -3.0e38f;
  for (int i = 0; i < 256; ++i)
    mx = fmaxf(mx, fm4[((size_t)b * 256 + i) * 512 + c]);
  fg[b * 512 + c] = mx;
}

// ---------------- f32 -> bf16 converters (pad K to Kp) ----------------
__global__ void k_cvtA(const float* __restrict__ src, unsigned short* __restrict__ dst,
                       int K, int Kp, size_t total) {
  size_t id = (size_t)blockIdx.x * blockDim.x + threadIdx.x;
  if (id >= total) return;
  size_t row = id / Kp; int kc = (int)(id % Kp);
  dst[id] = (kc < K) ? f2bf(src[row * K + kc]) : (unsigned short)0;
}
// W [K][Nn] -> dst [Nn][Kp]  (transpose)
__global__ void k_cvtWT(const float* __restrict__ W, unsigned short* __restrict__ dst,
                        int K, int Kp, int Nn, size_t total) {
  size_t id = (size_t)blockIdx.x * blockDim.x + threadIdx.x;
  if (id >= total) return;
  int n = (int)(id / Kp); int kc = (int)(id % Kp);
  dst[id] = (kc < K) ? f2bf(W[(size_t)kc * Nn + n]) : (unsigned short)0;
}
// W [O][K] -> dst [O][Kp]   (pconv weights already out-major)
__global__ void k_cvtWD(const float* __restrict__ W, unsigned short* __restrict__ dst,
                        int K, int Kp, size_t total) {
  size_t id = (size_t)blockIdx.x * blockDim.x + threadIdx.x;
  if (id >= total) return;
  size_t o = id / Kp; int kc = (int)(id % Kp);
  dst[id] = (kc < K) ? f2bf(W[o * K + kc]) : (unsigned short)0;
}

// =====================================================================
// Block-cooperative WMMA bf16 GEMM:
//   C[M][Nn] = A[M][Kp] * Bt[Nn][Kp]^T + bias
// Block = 256 thr = 8 waves (2 x 4), macro-tile 64 x 128.
// Each wave owns a 2x2 grid of 16x16 f32 accumulators.
// A/B K-slices staged to LDS with GLOBAL_LOAD_ASYNC_TO_LDS_B128
// (ASYNCcnt), double-buffered; fragments read back via ds_load_b128.
// =====================================================================
__global__ void __launch_bounds__(256)
k_gemm_bf16(const unsigned short* __restrict__ A, const unsigned short* __restrict__ Bt,
            const float* __restrict__ bias, float* __restrict__ C,
            int M, int Kp, int Nn) {
  __shared__ __align__(16) unsigned short As[2][64 * 32];    // 4KB per buf
  __shared__ __align__(16) unsigned short Bs[2][128 * 32];   // 8KB per buf
  const int tid  = threadIdx.x;
  const int wid  = tid >> 5;           // 0..7
  const int lane = tid & 31;
  const int wm   = wid >> 2;           // 0..1   (M direction)
  const int wn   = wid & 3;            // 0..3   (N direction)
  const int r    = lane & 15, hi = lane >> 4;
  const int ca   = hi ? 8 : 0;         // A frag: lanes16-31 hold K 8..15 / 24..31
  const int cb   = hi ? 8 : 0;         // B frag: lanes16-31 hold K 8..15 / 24..31
  const int rowBlk = blockIdx.y * 64;
  const int colBlk = blockIdx.x * 128;

  const char* gA = (const char*)(A  + (size_t)rowBlk * Kp);
  const char* gB = (const char*)(Bt + (size_t)colBlk * Kp);
  const unsigned ldsA = (unsigned)(uintptr_t)(&As[0][0]);
  const unsigned ldsB = (unsigned)(uintptr_t)(&Bs[0][0]);
  const int strideB = Kp * 2;          // bytes per row (both A and Bt)

  // per-lane async-copy chunks: tile rows are 64 bytes (32 bf16) wide
  const int eA  = wid * 512 + lane * 16;            // A tile: 4096B, 1 chunk/wave
  const int rA  = eA >> 6,  cAo = eA & 63;
  const int eB0 = (2 * wid) * 512 + lane * 16;      // B tile: 8192B, 2 chunks/wave
  const int eB1 = eB0 + 512;
  const int rB0 = eB0 >> 6, cB0 = eB0 & 63;
  const int rB1 = eB1 >> 6, cB1 = eB1 & 63;

  auto issue = [&](int buf, int k0) {
    int kb = k0 * 2;                                // byte offset along K
    async_copy_b128(ldsA + (unsigned)(buf * 64 * 32 * 2) + (unsigned)eA,
                    (unsigned long long)(uintptr_t)(gA + (size_t)rA * strideB + kb + cAo));
    async_copy_b128(ldsB + (unsigned)(buf * 128 * 32 * 2) + (unsigned)eB0,
                    (unsigned long long)(uintptr_t)(gB + (size_t)rB0 * strideB + kb + cB0));
    async_copy_b128(ldsB + (unsigned)(buf * 128 * 32 * 2) + (unsigned)eB1,
                    (unsigned long long)(uintptr_t)(gB + (size_t)rB1 * strideB + kb + cB1));
    __builtin_prefetch(gA + (size_t)rA * strideB + kb + cAo + 64, 0, 1);   // warm next slice
    __builtin_prefetch(gB + (size_t)rB0 * strideB + kb + cB0 + 64, 0, 1);
  };

  v8f acc[2][2];
#pragma unroll
  for (int i = 0; i < 2; ++i)
#pragma unroll
    for (int j = 0; j < 2; ++j)
#pragma unroll
      for (int t = 0; t < 8; ++t) acc[i][j][t] = 0.0f;

  issue(0, 0);
  int buf = 0;
  for (int k0 = 0; k0 < Kp; k0 += 32) {
    wait_async0();               // own chunks landed in LDS
    __syncthreads();             // everyone's chunks landed
    if (k0 + 32 < Kp) issue(buf ^ 1, k0 + 32);

    union { v16bf v; u32x4 u[2]; } af[2], bfr[2];
#pragma unroll
    for (int i = 0; i < 2; ++i) {
      const unsigned short* ap = &As[buf][(wm * 32 + i * 16 + r) * 32];
      af[i].u[0] = *(const u32x4*)(ap + ca);
      af[i].u[1] = *(const u32x4*)(ap + ca + 16);
    }
#pragma unroll
    for (int j = 0; j < 2; ++j) {
      const unsigned short* bp = &Bs[buf][(wn * 32 + j * 16 + r) * 32];
      bfr[j].u[0] = *(const u32x4*)(bp + cb);
      bfr[j].u[1] = *(const u32x4*)(bp + cb + 16);
    }
#pragma unroll
    for (int i = 0; i < 2; ++i)
#pragma unroll
      for (int j = 0; j < 2; ++j)
        acc[i][j] = __builtin_amdgcn_wmma_f32_16x16x32_bf16(false, af[i].v, false, bfr[j].v,
                                                            (short)0, acc[i][j], false, false);
    __syncthreads();             // all reads of buf done before it is refilled
    buf ^= 1;
  }

#pragma unroll
  for (int j = 0; j < 2; ++j) {
    int col = colBlk + wn * 32 + j * 16 + r;
    float bv = bias[col];
#pragma unroll
    for (int i = 0; i < 2; ++i) {
      int rowB = rowBlk + wm * 32 + i * 16 + ca;   // C/D layout: M = 8*half + vgpr
#pragma unroll
      for (int t = 0; t < 8; ++t)
        C[(size_t)(rowB + t) * Nn + col] = acc[i][j][t] + bv;
    }
  }
}

// ---------------- build feat (1286 -> pad 1312) directly in bf16 ----------------
__global__ void k_build_feat(const float* __restrict__ fm0, const float* __restrict__ fm1,
                             const float* __restrict__ fm2, const float* __restrict__ fm3,
                             const float* __restrict__ fm4, const int* __restrict__ n1,
                             const int* __restrict__ n2, const int* __restrict__ cat,
                             unsigned short* __restrict__ abf) {
  int p = blockIdx.x; int b = p >> 12;
  int r1 = b * 1024 + n1[p];
  int r2 = b * 256  + n2[p];
  int ct = cat[b];
  for (int j = threadIdx.x; j < 1312; j += blockDim.x) {
    float v;
    if      (j < 128)  v = fm0[(size_t)p * 128 + j];
    else if (j < 256)  v = fm1[(size_t)p * 128 + (j - 128)];
    else if (j < 512)  v = fm2[(size_t)r1 * 256 + (j - 256)];
    else if (j < 768)  v = fm3[(size_t)r1 * 256 + (j - 512)];
    else if (j < 1280) v = fm4[(size_t)r2 * 512 + (j - 768)];
    else if (j < 1286) v = ((j - 1280) == ct) ? 1.0f : 0.0f;
    else               v = 0.0f;
    abf[(size_t)p * 1312 + j] = f2bf(v);
  }
}

// ---------------- build face_in (771 -> pad 800) in bf16 ----------------
__global__ void k_build_facein(const float* __restrict__ fg, const float* __restrict__ h3,
                               const float* __restrict__ V, unsigned short* __restrict__ abf) {
  int p = blockIdx.x; int b = p >> 12;
  for (int j = threadIdx.x; j < 800; j += blockDim.x) {
    float v;
    if      (j < 512) v = fg[b * 512 + j];
    else if (j < 768) v = h3[(size_t)p * 256 + (j - 512)];
    else if (j < 771) v = V[(size_t)p * 3 + (j - 768)];
    else              v = 0.0f;
    abf[(size_t)p * 800 + j] = f2bf(v);
  }
}

// ---------------- final: kw4 pconv at n=0 only, + center ----------------
__global__ void k_final(const float* __restrict__ kk3, const float* __restrict__ kw4,
                        const float* __restrict__ kb4, const float* __restrict__ center,
                        float* __restrict__ out) {
  int t = threadIdx.x;
  if (t >= 48) return;
  int b = t / 24, r = t % 24;
  const float* x = kk3 + (size_t)(b * 4096) * 128;   // point n=0 of batch b
  float acc = kb4[r];
  for (int c = 0; c < 128; ++c) acc += kw4[r * 128 + c] * x[c];
  out[t] = acc + center[b * 3 + (r % 3)];
}

// =====================================================================
extern "C" void kernel_launch(void* const* d_in, const int* in_sizes, int n_in,
                              void* d_out, int out_size, void* d_ws, size_t ws_size,
                              hipStream_t stream) {
  (void)in_sizes; (void)n_in; (void)out_size; (void)ws_size;
  const float* pts = (const float*)d_in[0];
  const int*   cat = (const int*)d_in[1];
  const float* d0  = (const float*)d_in[2];
  const float* w1  = (const float*)d_in[3];
  const float* b1  = (const float*)d_in[4];
  const float* d1  = (const float*)d_in[5];
  const float* w2  = (const float*)d_in[6];
  const float* b2  = (const float*)d_in[7];
  const float* d2  = (const float*)d_in[8];
  const float* w3  = (const float*)d_in[9];
  const float* b3  = (const float*)d_in[10];
  const float* d3  = (const float*)d_in[11];
  const float* w4  = (const float*)d_in[12];
  const float* b4  = (const float*)d_in[13];
  const float* d4  = (const float*)d_in[14];
  const float* bn1g = (const float*)d_in[15]; const float* bn1b = (const float*)d_in[16];
  const float* bn2g = (const float*)d_in[17]; const float* bn2b = (const float*)d_in[18];
  const float* bn3g = (const float*)d_in[19]; const float* bn3b = (const float*)d_in[20];
  const float* cw1 = (const float*)d_in[21]; const float* cb1 = (const float*)d_in[22];
  const float* cg1 = (const float*)d_in[23]; const float* cbb1 = (const float*)d_in[24];
  const float* cw2 = (const float*)d_in[25]; const float* cb2 = (const float*)d_in[26];
  const float* cg2 = (const float*)d_in[27]; const float* cbb2 = (const float*)d_in[28];
  const float* cw3 = (const float*)d_in[29]; const float* cb3 = (const float*)d_in[30];
  const float* cg3 = (const float*)d_in[31]; const float* cbb3 = (const float*)d_in[32];
  const float* kw1 = (const float*)d_in[33]; const float* kb1 = (const float*)d_in[34];
  const float* kg1 = (const float*)d_in[35]; const float* kbb1 = (const float*)d_in[36];
  const float* kw2 = (const float*)d_in[37]; const float* kb2 = (const float*)d_in[38];
  const float* kg2 = (const float*)d_in[39]; const float* kbb2 = (const float*)d_in[40];
  const float* kw3 = (const float*)d_in[41]; const float* kb3 = (const float*)d_in[42];
  const float* kg3 = (const float*)d_in[43]; const float* kbb3 = (const float*)d_in[44];
  const float* kw4 = (const float*)d_in[45]; const float* kb4 = (const float*)d_in[46];

  // -------- workspace layout --------
  char* base = (char*)d_ws;
  size_t off = 0;
  auto alloc = [&](size_t bytes) -> char* {
    char* p = base + off;
    off = (off + bytes + 255) & ~(size_t)255;
    return p;
  };
  float* CENTER = (float*)alloc(2 * 3 * 4);
  float* V      = (float*)alloc((size_t)8192 * 3 * 4);
  float* SD     = (float*)alloc((size_t)26880 * 4);   // 3*(896+896+1792+1792+3584)
  int*   IDX0   = (int*)alloc((size_t)8192 * 10 * 4);
  int*   IDX1   = (int*)alloc((size_t)2048 * 10 * 4);
  int*   IDX2   = (int*)alloc((size_t)512 * 10 * 4);
  int*   IDXP1  = (int*)alloc((size_t)2048 * 4 * 4);
  int*   IDXP2  = (int*)alloc((size_t)512 * 4 * 4);
  int*   N1     = (int*)alloc((size_t)8192 * 4);
  int*   N2     = (int*)alloc((size_t)8192 * 4);
  float* FM0    = (float*)alloc((size_t)8192 * 128 * 4);
  float* FM1    = (float*)alloc((size_t)8192 * 128 * 4);
  float* FM2    = (float*)alloc((size_t)2048 * 256 * 4);
  float* FM3    = (float*)alloc((size_t)2048 * 256 * 4);
  float* FM4    = (float*)alloc((size_t)512 * 512 * 4);
  float* FP1    = (float*)alloc((size_t)2048 * 128 * 4);
  float* FP2    = (float*)alloc((size_t)512 * 256 * 4);
  float* FGLOB  = (float*)alloc((size_t)2 * 512 * 4);
  float* BNS    = (float*)alloc((size_t)1024 * 4);
  float* GOUT   = (float*)alloc((size_t)8192 * 1024 * 4);
  unsigned short* ABF = (unsigned short*)alloc((size_t)8192 * 1312 * 2);
  unsigned short* WBF = (unsigned short*)alloc((size_t)4096 * 256 * 2);
  float* H1 = (float*)alloc((size_t)8192 * 512 * 4);   // later KK1
  float* H2 = (float*)alloc((size_t)8192 * 512 * 4);   // later KK2
  float* H3 = (float*)alloc((size_t)8192 * 256 * 4);   // later KK3

  auto cdiv = [](size_t a, size_t b) -> unsigned { return (unsigned)((a + b - 1) / b); };
  auto gemm = [&](const unsigned short* Ab, const unsigned short* Wb, const float* bias,
                  float* Cout, int M, int Kp, int Nn) {
    k_gemm_bf16<<<dim3(Nn / 128, M / 64), dim3(256), 0, stream>>>(Ab, Wb, bias, Cout, M, Kp, Nn);
  };
  auto cvtA = [&](const float* s, int K, int Kp, size_t rows) {
    size_t tot = rows * (size_t)Kp;
    k_cvtA<<<dim3(cdiv(tot, 256)), dim3(256), 0, stream>>>(s, ABF, K, Kp, tot);
  };

  // SD offsets (floats)
  float* SD0 = SD + 0;      // 3 x 896
  float* SD1 = SD + 2688;   // 3 x 896
  float* SD2 = SD + 5376;   // 3 x 1792
  float* SD3 = SD + 10752;  // 3 x 1792
  float* SD4 = SD + 16128;  // 3 x 3584

  // -------- geometry --------
  k_center_v<<<dim3(2), dim3(256), 0, stream>>>(pts, CENTER, V);
  k_normdir<<<dim3(cdiv(896, 256)), dim3(256), 0, stream>>>(d0, SD0, 896);
  k_normdir<<<dim3(cdiv(896, 256)), dim3(256), 0, stream>>>(d1, SD1, 896);
  k_normdir<<<dim3(cdiv(1792, 256)), dim3(256), 0, stream>>>(d2, SD2, 1792);
  k_normdir<<<dim3(cdiv(1792, 256)), dim3(256), 0, stream>>>(d3, SD3, 1792);
  k_normdir<<<dim3(cdiv(3584, 256)), dim3(256), 0, stream>>>(d4, SD4, 3584);
  k_knn<10><<<dim3(cdiv(8192, 128)), dim3(128), 0, stream>>>(V, 4096, 4096, IDX0, 8192);

  // -------- fm0 --------
  k_conv_surface<<<dim3(8192), dim3(128), 0, stream>>>(V, IDX0, SD0, FM0, 4096);

  // -------- fm1 --------
  cvtA(FM0, 128, 128, 8192);
  k_cvtWT<<<dim3(cdiv((size_t)1024 * 128, 256)), dim3(256), 0, stream>>>(w1, WBF, 128, 128, 1024, (size_t)1024 * 128);
  gemm(ABF, WBF, b1, GOUT, 8192, 128, 1024);
  k_conv_agg<<<dim3(8192), dim3(128), 0, stream>>>(V, IDX0, SD1, GOUT, FM1, 4096, 128);
  k_bn_stats<<<dim3(128), dim3(256), 0, stream>>>(FM1, BNS, 8192, 128);
  k_bn_apply<<<dim3(cdiv((size_t)8192 * 128, 256)), dim3(256), 0, stream>>>(FM1, FM1, bn1g, bn1b, BNS, (size_t)8192 * 128, 128);

  // -------- pool1, idx1 --------
  k_knn<4><<<dim3(cdiv(2048, 128)), dim3(128), 0, stream>>>(V, 1024, 4096, IDXP1, 2048);
  k_gather_max<<<dim3(2048), dim3(128), 0, stream>>>(FM1, IDXP1, FP1, 128, 1024, 4096);
  k_knn<10><<<dim3(cdiv(2048, 128)), dim3(128), 0, stream>>>(V, 1024, 1024, IDX1, 2048);

  // -------- fm2 --------
  cvtA(FP1, 128, 128, 2048);
  k_cvtWT<<<dim3(cdiv((size_t)2048 * 128, 256)), dim3(256), 0, stream>>>(w2, WBF, 128, 128, 2048, (size_t)2048 * 128);
  gemm(ABF, WBF, b2, GOUT, 2048, 128, 2048);
  k_conv_agg<<<dim3(2048), dim3(256), 0, stream>>>(V, IDX1, SD2, GOUT, FM2, 1024, 256);
  k_bn_stats<<<dim3(256), dim3(256), 0, stream>>>(FM2, BNS, 2048, 256);
  k_bn_apply<<<dim3(cdiv((size_t)2048 * 256, 256)), dim3(256), 0, stream>>>(FM2, FM2, bn2g, bn2b, BNS, (size_t)2048 * 256, 256);

  // -------- fm3 --------
  cvtA(FM2, 256, 256, 2048);
  k_cvtWT<<<dim3(cdiv((size_t)2048 * 256, 256)), dim3(256), 0, stream>>>(w3, WBF, 256, 256, 2048, (size_t)2048 * 256);
  gemm(ABF, WBF, b3, GOUT, 2048, 256, 2048);
  k_conv_agg<<<dim3(2048), dim3(256), 0, stream>>>(V, IDX1, SD3, GOUT, FM3, 1024, 256);
  k_bn_stats<<<dim3(256), dim3(256), 0, stream>>>(FM3, BNS, 2048, 256);
  k_bn_apply<<<dim3(cdiv((size_t)2048 * 256, 256)), dim3(256), 0, stream>>>(FM3, FM3, bn3g, bn3b, BNS, (size_t)2048 * 256, 256);

  // -------- pool2, idx2 --------
  k_knn<4><<<dim3(cdiv(512, 128)), dim3(128), 0, stream>>>(V, 256, 1024, IDXP2, 512);
  k_gather_max<<<dim3(512), dim3(256), 0, stream>>>(FM3, IDXP2, FP2, 256, 256, 1024);
  k_knn<10><<<dim3(cdiv(512, 128)), dim3(128), 0, stream>>>(V, 256, 256, IDX2, 512);

  // -------- fm4 (no BN) --------
  cvtA(FP2, 256, 256, 512);
  k_cvtWT<<<dim3(cdiv((size_t)4096 * 256, 256)), dim3(256), 0, stream>>>(w4, WBF, 256, 256, 4096, (size_t)4096 * 256);
  gemm(ABF, WBF, b4, GOUT, 512, 256, 4096);
  k_conv_agg<<<dim3(512), dim3(512), 0, stream>>>(V, IDX2, SD4, GOUT, FM4, 256, 512);
  k_global_max<<<dim3(2), dim3(512), 0, stream>>>(FM4, FGLOB);

  // -------- nearest upsample indices --------
  k_nearest<<<dim3(cdiv(8192, 128)), dim3(128), 0, stream>>>(V, 1024, N1);
  k_nearest<<<dim3(cdiv(8192, 128)), dim3(128), 0, stream>>>(V, 256, N2);

  // -------- seg head: cw1/cw2/cw3 --------
  k_build_feat<<<dim3(8192), dim3(256), 0, stream>>>(FM0, FM1, FM2, FM3, FM4, N1, N2, cat, ABF);
  k_cvtWD<<<dim3(cdiv((size_t)512 * 1312, 256)), dim3(256), 0, stream>>>(cw1, WBF, 1286, 1312, (size_t)512 * 1312);
  gemm(ABF, WBF, cb1, GOUT, 8192, 1312, 512);
  k_bn_stats<<<dim3(512), dim3(256), 0, stream>>>(GOUT, BNS, 8192, 512);
  k_bn_apply<<<dim3(cdiv((size_t)8192 * 512, 256)), dim3(256), 0, stream>>>(GOUT, H1, cg1, cbb1, BNS, (size_t)8192 * 512, 512);

  cvtA(H1, 512, 512, 8192);
  k_cvtWD<<<dim3(cdiv((size_t)512 * 512, 256)), dim3(256), 0, stream>>>(cw2, WBF, 512, 512, (size_t)512 * 512);
  gemm(ABF, WBF, cb2, GOUT, 8192, 512, 512);
  k_bn_stats<<<dim3(512), dim3(256), 0, stream>>>(GOUT, BNS, 8192, 512);
  k_bn_apply<<<dim3(cdiv((size_t)8192 * 512, 256)), dim3(256), 0, stream>>>(GOUT, H2, cg2, cbb2, BNS, (size_t)8192 * 512, 512);

  cvtA(H2, 512, 512, 8192);
  k_cvtWD<<<dim3(cdiv((size_t)256 * 512, 256)), dim3(256), 0, stream>>>(cw3, WBF, 512, 512, (size_t)256 * 512);
  gemm(ABF, WBF, cb3, GOUT, 8192, 512, 256);
  k_bn_stats<<<dim3(256), dim3(256), 0, stream>>>(GOUT, BNS, 8192, 256);
  k_bn_apply<<<dim3(cdiv((size_t)8192 * 256, 256)), dim3(256), 0, stream>>>(GOUT, H3, cg3, cbb3, BNS, (size_t)8192 * 256, 256);

  // -------- key head: kw1/kw2/kw3 (KK1->H1, KK2->H2, KK3->H3) --------
  k_build_facein<<<dim3(8192), dim3(256), 0, stream>>>(FGLOB, H3, V, ABF);
  k_cvtWD<<<dim3(cdiv((size_t)512 * 800, 256)), dim3(256), 0, stream>>>(kw1, WBF, 771, 800, (size_t)512 * 800);
  gemm(ABF, WBF, kb1, GOUT, 8192, 800, 512);
  k_bn_stats<<<dim3(512), dim3(256), 0, stream>>>(GOUT, BNS, 8192, 512);
  k_bn_apply<<<dim3(cdiv((size_t)8192 * 512, 256)), dim3(256), 0, stream>>>(GOUT, H1, kg1, kbb1, BNS, (size_t)8192 * 512, 512);

  cvtA(H1, 512, 512, 8192);
  k_cvtWD<<<dim3(cdiv((size_t)256 * 512, 256)), dim3(256), 0, stream>>>(kw2, WBF, 512, 512, (size_t)256 * 512);
  gemm(ABF, WBF, kb2, GOUT, 8192, 512, 256);
  k_bn_stats<<<dim3(256), dim3(256), 0, stream>>>(GOUT, BNS, 8192, 256);
  k_bn_apply<<<dim3(cdiv((size_t)8192 * 256, 256)), dim3(256), 0, stream>>>(GOUT, H2, kg2, kbb2, BNS, (size_t)8192 * 256, 256);

  cvtA(H2, 256, 256, 8192);
  k_cvtWD<<<dim3(cdiv((size_t)128 * 256, 256)), dim3(256), 0, stream>>>(kw3, WBF, 256, 256, (size_t)128 * 256);
  gemm(ABF, WBF, kb3, GOUT, 8192, 256, 128);
  k_bn_stats<<<dim3(128), dim3(256), 0, stream>>>(GOUT, BNS, 8192, 128);
  k_bn_apply<<<dim3(cdiv((size_t)8192 * 128, 256)), dim3(256), 0, stream>>>(GOUT, H3, kg3, kbb3, BNS, (size_t)8192 * 128, 128);

  // -------- final output (2,8,3) --------
  k_final<<<dim3(1), dim3(64), 0, stream>>>(H3, kw4, kb4, CENTER, (float*)d_out);
}
